// MultiHeadAttn_19619410608618
// MI455X (gfx1250) — matrix operand order repaired and checked by reading, not compiled, not run on hardware
//
#include <hip/hip_runtime.h>
#include <hip/hip_bf16.h>
#include <math.h>

// CDNA5 (gfx1250) wave32 WMMA implementation of Transformer-XL MHA block:
//   qkv = h @ Wqkv               (bf16 WMMA GEMM, f32 accum)
//   flash-attention per (b,head) (bf16 WMMA, online softmax in f32)
//   attn_out = attn_vec @ Wo     (bf16 WMMA GEMM, f32 accum)
//   out = LayerNorm(h + attn_out)

typedef __attribute__((ext_vector_type(16))) __bf16 v16bf;
typedef __attribute__((ext_vector_type(8)))  float  v8f;

#define S_LEN  2048
#define BATCH  2
#define DMODEL 1024
#define NHEAD  16
#define DHEAD  64
#define ROWS   (S_LEN*BATCH)   // 4096
#define QKV_N  (3*DMODEL)      // 3072

static __device__ inline unsigned lds_pair(const __bf16* p) {
  // two consecutive bf16 (even K, K+1) as one 32-bit LDS read
  return *(const unsigned*)p;
}

// A-matrix fragment, 16x32 bf16 (this code's k order, not the ISA table): lane L: M = L&15,
// kb = (L<16)?0:8; VGPR i: K = 2*kb+2i (contiguous K-half, matches B).
static __device__ inline v16bf load_a_frag(const __bf16* rowbase, int half) {
  union { v16bf v; unsigned u[8]; } f;
  const int kb = half * 8;
#pragma unroll
  for (int i = 0; i < 8; ++i) {
    const int k = 2*kb + 2*i;
    f.u[i] = lds_pair(rowbase + k);
  }
  return f.v;
}

// B-matrix fragment, 32x16 bf16: lane L: N = L&15, K-half = (L<16)?0:16,
// VGPR i: K = Khalf + 2i. Caller passes pointer already offset by K-half;
// data must be laid out K-contiguous per column (we store B transposed).
static __device__ inline v16bf load_b_frag(const __bf16* colbase) {
  union { v16bf v; unsigned u[8]; } f;
#pragma unroll
  for (int i = 0; i < 8; ++i) f.u[i] = lds_pair(colbase + 2*i);
  return f.v;
}

static __device__ inline v8f vzero8() {
  v8f z = {0.f,0.f,0.f,0.f,0.f,0.f,0.f,0.f};
  return z;
}

// -------------------------------------------------------------------------
// Tiled bf16 WMMA GEMM: C[M,N] = A[M,K] * B[K,N].
// Block = 256 threads (8 waves), tile 128x128, K-step 32 (one WMMA deep).
// Wave grid 4(M) x 2(N): each wave computes 32x64 = 2x4 16x16 frags.
// -------------------------------------------------------------------------
template <typename AT, typename OT>
__global__ __launch_bounds__(256) void gemm_bf16_wmma(
    const AT* __restrict__ A, const float* __restrict__ B, OT* __restrict__ C,
    int M, int N, int K)
{
  __shared__ __bf16 ldsA[128*32];   // [row][k]   row-major
  __shared__ __bf16 ldsBT[128*32];  // [col][k]   (B transposed -> K contiguous)

  const int tid  = threadIdx.x;
  const int wid  = tid >> 5;
  const int lane = tid & 31;
  const int half = lane >> 4;
  const int l16  = lane & 15;
  const int wm   = wid & 3;       // wave row tile  (rows wm*32 .. +31)
  const int wn   = wid >> 2;      // wave col tile  (cols wn*64 .. +63)
  const int mBase = blockIdx.y * 128;
  const int nBase = blockIdx.x * 128;

  v8f acc[2][4];
#pragma unroll
  for (int i = 0; i < 2; ++i)
#pragma unroll
    for (int j = 0; j < 4; ++j) acc[i][j] = vzero8();

  for (int k0 = 0; k0 < K; k0 += 32) {
    // Stage A tile 128x32 (fp32/bf16 -> bf16), 16 elems/thread, coalesced.
#pragma unroll
    for (int i = 0; i < 16; ++i) {
      int idx = tid + i*256;
      int r = idx >> 5, c = idx & 31;
      ldsA[idx] = (__bf16)A[(size_t)(mBase + r)*K + k0 + c];
    }
    // Stage B tile 32x128 transposed into [n][k].
#pragma unroll
    for (int i = 0; i < 16; ++i) {
      int idx = tid + i*256;
      int kk = idx >> 7, nn = idx & 127;
      ldsBT[nn*32 + kk] = (__bf16)B[(size_t)(k0 + kk)*N + nBase + nn];
    }
    __syncthreads();

    v16bf afr[2], bfr[4];
#pragma unroll
    for (int fm = 0; fm < 2; ++fm)
      afr[fm] = load_a_frag(&ldsA[(wm*32 + fm*16 + l16)*32], half);
#pragma unroll
    for (int fn = 0; fn < 4; ++fn)
      bfr[fn] = load_b_frag(&ldsBT[(wn*64 + fn*16 + l16)*32 + half*16]);

#pragma unroll
    for (int fm = 0; fm < 2; ++fm)
#pragma unroll
      for (int fn = 0; fn < 4; ++fn)
        acc[fm][fn] = __builtin_amdgcn_wmma_f32_16x16x32_bf16(
            false, afr[fm], false, bfr[fn], (short)0, acc[fm][fn], false, false);
    __syncthreads();
  }

  // C/D layout: lane N = l16, VGPR r -> M = r + half*8.
#pragma unroll
  for (int fm = 0; fm < 2; ++fm)
#pragma unroll
    for (int fn = 0; fn < 4; ++fn)
#pragma unroll
      for (int r = 0; r < 8; ++r) {
        int row = mBase + wm*32 + fm*16 + r + half*8;
        int col = nBase + wn*64 + fn*16 + l16;
        C[(size_t)row*N + col] = (OT)acc[fm][fn][r];
      }
}

// -------------------------------------------------------------------------
// Flash attention. Block = (q-tile of 128, one (batch,head)), 8 waves.
// Wave w owns query rows q0+w*16 .. +15. Key tiles of 64, online softmax.
// qkv layout: row (s*B+b), cols [0,1024)=Q, [1024,2048)=K, [2048,3072)=V,
// head n occupies cols n*64..+63 of each third (matches reference reshape).
// -------------------------------------------------------------------------
__global__ __launch_bounds__(256) void attn_flash(
    const __bf16* __restrict__ qkv, __bf16* __restrict__ avec)
{
  __shared__ __bf16 ldsQ[128*64];     // [qrow][d]
  __shared__ __bf16 ldsK[64*64];      // [key][d]  (K-frag B: K-dim = d, contiguous)
  __shared__ __bf16 ldsVT[64*64];     // [d][key]  (V-frag B: K-dim = key, contiguous)
  __shared__ __bf16 ldsP[8][16*64];   // per-wave P spill (C-layout -> A-layout transpose)

  const int tid  = threadIdx.x;
  const int w    = tid >> 5;
  const int lane = tid & 31;
  const int half = lane >> 4;
  const int l16  = lane & 15;
  const int bh = blockIdx.y;
  const int b  = bh / NHEAD;
  const int n  = bh % NHEAD;
  const int q0 = blockIdx.x * 128;
  const int hoff = n * DHEAD;

  // Stage Q tile once.
#pragma unroll
  for (int i = 0; i < 32; ++i) {
    int idx = tid + i*256;
    int r = idx >> 6, d = idx & 63;
    ldsQ[idx] = qkv[((size_t)(q0 + r)*BATCH + b)*QKV_N + hoff + d];
  }
  __syncthreads();

  v16bf qfr[2];  // Dh = 64 -> two K=32 WMMA steps
#pragma unroll
  for (int ks = 0; ks < 2; ++ks)
    qfr[ks] = load_a_frag(&ldsQ[(w*16 + l16)*64 + ks*32], half);

  v8f o[4];      // 16 rows x 64 d accumulator
#pragma unroll
  for (int c = 0; c < 4; ++c) o[c] = vzero8();
  float m_run[8], l_run[8];
#pragma unroll
  for (int r = 0; r < 8; ++r) { m_run[r] = -3.0e38f; l_run[r] = 0.f; }

  const int qmax_w = q0 + w*16 + 15;
  const int nkt = 2*blockIdx.x + 2;   // key tiles needed under causality

  for (int t = 0; t < nkt; ++t) {
    const int j0 = t * 64;
    // Stage K (row-major) and V (transposed) tiles.
#pragma unroll
    for (int i = 0; i < 16; ++i) {
      int idx = tid + i*256;
      int kr = idx >> 6, d = idx & 63;
      size_t rb = ((size_t)(j0 + kr)*BATCH + b)*QKV_N + hoff;
      ldsK[idx]        = qkv[rb + DMODEL + d];
      ldsVT[d*64 + kr] = qkv[rb + 2*DMODEL + d];
    }
    __syncthreads();

    if (j0 <= qmax_w) {   // skip fully-masked tiles for this wave
      // S = Q K^T : 4 frags of 16 keys each.
      v8f s[4];
#pragma unroll
      for (int fj = 0; fj < 4; ++fj) {
        v8f sf = vzero8();
#pragma unroll
        for (int ks = 0; ks < 2; ++ks) {
          v16bf kfr = load_b_frag(&ldsK[(fj*16 + l16)*64 + ks*32 + half*16]);
          sf = __builtin_amdgcn_wmma_f32_16x16x32_bf16(
              false, qfr[ks], false, kfr, (short)0, sf, false, false);
        }
        s[fj] = sf;
      }
      // Scale 1/sqrt(64), causal mask, per-row max (16-lane half reduction).
      float mnew[8];
#pragma unroll
      for (int r = 0; r < 8; ++r) {
        const int qr = q0 + w*16 + half*8 + r;
        float mx = -3.0e38f;
#pragma unroll
        for (int fj = 0; fj < 4; ++fj) {
          float v = (float)s[fj][r] * 0.125f;
          if (j0 + fj*16 + l16 > qr) v = -3.0e38f;
          s[fj][r] = v;
          mx = fmaxf(mx, v);
        }
#pragma unroll
        for (int off = 1; off < 16; off <<= 1)
          mx = fmaxf(mx, __shfl_xor(mx, off, 32));
        mnew[r] = fmaxf(m_run[r], mx);
      }
      // Rescale running state, exponentiate, spill P (bf16) to LDS.
      float rowsum[8];
#pragma unroll
      for (int r = 0; r < 8; ++r) {
        float alpha = __expf(m_run[r] - mnew[r]);
        m_run[r] = mnew[r];
        l_run[r] *= alpha;
#pragma unroll
        for (int c = 0; c < 4; ++c) o[c][r] = (float)o[c][r] * alpha;
        rowsum[r] = 0.f;
      }
#pragma unroll
      for (int fj = 0; fj < 4; ++fj)
#pragma unroll
        for (int r = 0; r < 8; ++r) {
          float p = __expf((float)s[fj][r] - m_run[r]);
          rowsum[r] += p;
          ldsP[w][(half*8 + r)*64 + fj*16 + l16] = (__bf16)p;
        }
#pragma unroll
      for (int r = 0; r < 8; ++r) {
#pragma unroll
        for (int off = 1; off < 16; off <<= 1)
          rowsum[r] += __shfl_xor(rowsum[r], off, 32);
        l_run[r] += rowsum[r];
      }
      // O += P @ V  (K-dim = 64 keys -> 2 WMMA steps; per-wave LDS, in-order DS).
#pragma unroll
      for (int ks = 0; ks < 2; ++ks) {
        v16bf pfr = load_a_frag(&ldsP[w][l16*64 + ks*32], half);
#pragma unroll
        for (int c = 0; c < 4; ++c) {
          v16bf vfr = load_b_frag(&ldsVT[(c*16 + l16)*64 + ks*32 + half*16]);
          o[c] = __builtin_amdgcn_wmma_f32_16x16x32_bf16(
              false, pfr, false, vfr, (short)0, o[c], false, false);
        }
      }
    }
    __syncthreads();   // before next tile overwrites ldsK/ldsVT
  }

  // Normalize and write attn_vec (bf16) [row = s*B+b, col = n*64+d].
#pragma unroll
  for (int r = 0; r < 8; ++r) {
    const float inv = 1.0f / l_run[r];
    const int qr = q0 + w*16 + half*8 + r;
#pragma unroll
    for (int c = 0; c < 4; ++c) {
      int d = c*16 + l16;
      avec[((size_t)qr*BATCH + b)*DMODEL + hoff + d] = (__bf16)((float)o[c][r] * inv);
    }
  }
}

// -------------------------------------------------------------------------
// Residual + LayerNorm (biased variance). One block per row of 1024.
// -------------------------------------------------------------------------
__global__ __launch_bounds__(256) void resid_ln(
    const float* __restrict__ h, const float* __restrict__ ao,
    const float* __restrict__ gamma, const float* __restrict__ beta,
    float* __restrict__ out)
{
  __shared__ float red[8];
  const int row = blockIdx.x;
  const int tid = threadIdx.x;
  const int wid = tid >> 5, lane = tid & 31;
  const size_t base = (size_t)row * DMODEL;

  float x[4];
  float sum = 0.f;
#pragma unroll
  for (int i = 0; i < 4; ++i) {
    int c = tid + i*256;
    x[i] = h[base + c] + ao[base + c];
    sum += x[i];
  }
#pragma unroll
  for (int off = 16; off >= 1; off >>= 1) sum += __shfl_xor(sum, off, 32);
  if (lane == 0) red[wid] = sum;
  __syncthreads();
  sum = 0.f;
#pragma unroll
  for (int i = 0; i < 8; ++i) sum += red[i];
  const float mu = sum * (1.0f/DMODEL);

  float vs = 0.f;
#pragma unroll
  for (int i = 0; i < 4; ++i) { float d = x[i] - mu; vs += d*d; }
#pragma unroll
  for (int off = 16; off >= 1; off >>= 1) vs += __shfl_xor(vs, off, 32);
  __syncthreads();
  if (lane == 0) red[wid] = vs;
  __syncthreads();
  vs = 0.f;
#pragma unroll
  for (int i = 0; i < 8; ++i) vs += red[i];
  const float inv = rsqrtf(vs * (1.0f/DMODEL) + 1e-5f);

#pragma unroll
  for (int i = 0; i < 4; ++i) {
    int c = tid + i*256;
    out[base + c] = (x[i] - mu) * inv * gamma[c] + beta[c];
  }
}

extern "C" void kernel_launch(void* const* d_in, const int* in_sizes, int n_in,
                              void* d_out, int out_size, void* d_ws, size_t ws_size,
                              hipStream_t stream) {
  const float* h     = (const float*)d_in[0];
  // d_in[1] = attn_mask: pure causal triu -> computed analytically in-kernel.
  const float* Wqkv  = (const float*)d_in[2];
  const float* Wo    = (const float*)d_in[3];
  const float* gamma = (const float*)d_in[4];
  const float* beta  = (const float*)d_in[5];
  float* out = (float*)d_out;

  char* ws = (char*)d_ws;
  __bf16* qkv  = (__bf16*)ws;                                              // 4096x3072 bf16
  __bf16* avec = (__bf16*)(ws + (size_t)ROWS*QKV_N*2);                     // 4096x1024 bf16
  float*  aout = (float*)(ws + (size_t)ROWS*QKV_N*2 + (size_t)ROWS*DMODEL*2); // 4096x1024 f32

  // 1) QKV projection: [4096,1024] x [1024,3072] -> bf16 qkv scratch.
  gemm_bf16_wmma<float, __bf16>
      <<<dim3(QKV_N/128, ROWS/128), 256, 0, stream>>>(h, Wqkv, qkv, ROWS, QKV_N, DMODEL);

  // 2) Flash attention: 16 q-tiles x 32 (batch,head) blocks.
  attn_flash<<<dim3(S_LEN/128, BATCH*NHEAD), 256, 0, stream>>>(qkv, avec);

  // 3) Output projection: [4096,1024] x [1024,1024] -> f32 attn_out scratch.
  gemm_bf16_wmma<__bf16, float>
      <<<dim3(DMODEL/128, ROWS/128), 256, 0, stream>>>(avec, Wo, aout, ROWS, DMODEL, DMODEL);

  // 4) Residual + LayerNorm.
  resid_ln<<<ROWS, 256, 0, stream>>>(h, aout, gamma, beta, out);
}
